// DQN_42700564857292
// MI455X (gfx1250) — matrix operand order repaired and checked by reading, not compiled, not run on hardware
//
#include <hip/hip_runtime.h>
#include <hip/hip_bf16.h>

typedef __attribute__((ext_vector_type(16))) _Float16 v16h;
typedef __attribute__((ext_vector_type(8)))  _Float16 v8h;
typedef __attribute__((ext_vector_type(8)))  float    v8f;

union V16 { v16h v; v8h h2[2]; _Float16 e[16]; };

#define TSTEPS 512
#define BATCH  256
#define NLAYER 160

__device__ __forceinline__ float sigm(float x) {
  return __builtin_amdgcn_rcpf(1.0f + __expf(-x));
}
__device__ __forceinline__ float tanh_f(float x) {
  x = fminf(fmaxf(x, -15.0f), 15.0f);
  float e = __expf(2.0f * x);
  return (e - 1.0f) * __builtin_amdgcn_rcpf(e + 1.0f);
}
__device__ __forceinline__ void spin_ge(int* p, int target) {
  while (__hip_atomic_load(p, __ATOMIC_ACQUIRE, __HIP_MEMORY_SCOPE_AGENT) < target)
    __builtin_amdgcn_s_sleep(2);
}

// One workgroup == one LSTM layer. 8 waves x 32 batch rows each.
// K (= Din + H, padded to 32) maps exactly onto v_wmma_f32_16x16x32_f16.
template<int H>
__device__ __forceinline__ void run_layer(
    const float* __restrict__ Wih, const float* __restrict__ Whh,
    const float* __restrict__ bih, const float* __restrict__ bhh,
    int Din, int Hin, bool relu_in,
    const float* __restrict__ x0,
    const _Float16* __restrict__ in_ring,
    _Float16* __restrict__ out_ring,
    float* __restrict__ out_final,
    const float* fcls,          // LDS: fc_W (64) + fc_b (4)
    _Float16* zbuf,             // LDS: [256][32] f16 concat [x | h | pad]
    int* prog, int L, int wsteps)
{
  constexpr int NT = H / 4;     // N-tiles of 16 covering 4H gate columns
  const int tid   = threadIdx.x;
  const int lane  = tid & 31;
  const int wave  = tid >> 5;
  const int n     = lane & 15;
  const bool hig  = (lane >= 16);
  const int klo   = hig ? 8 : 0;    // A-matrix K grouping per ISA layout
  const int khalf = hig ? 16 : 0;   // unused; B is built with klo, same order as A
  const int wmask = wsteps - 1;

  // ---- Build resident B-fragments: W[k][g] = [Wih^T ; Whh^T], zero-padded to K=32
  V16 bfr[NT];
  #pragma unroll
  for (int nt = 0; nt < NT; ++nt) {
    int g = nt * 16 + n;
    #pragma unroll
    for (int j = 0; j < 16; ++j) {
      int k = (j < 8) ? (klo + j) : (16 + klo + (j - 8));
      float w = 0.0f;
      if (k < Din)          w = Wih[g * Din + k];
      else if (k < Din + H) w = Whh[g * H + (k - Din)];
      bfr[nt].e[j] = (_Float16)w;
    }
  }

  // ---- Per-lane gate biases (hh = n for owner lanes)
  int hb = (n < H) ? n : 0;
  float Bi = bih[0*H+hb] + bhh[0*H+hb];
  float Bf = bih[1*H+hb] + bhh[1*H+hb];
  float Bg = bih[2*H+hb] + bhh[2*H+hb];
  float Bo = bih[3*H+hb] + bhh[3*H+hb];

  // ---- Zero z (covers h0 = 0 and K padding); cell state lives in VGPRs
  {
    _Float16* zr = zbuf + tid * 32;
    #pragma unroll
    for (int k = 0; k < 32; ++k) zr[k] = (_Float16)0.0f;
  }
  float creg[2][8];
  #pragma unroll
  for (int m = 0; m < 2; ++m)
    #pragma unroll
    for (int r = 0; r < 8; ++r) creg[m][r] = 0.0f;
  __syncthreads();

  for (int t = 0; t < TSTEPS; ++t) {
    // ---- Pipeline flags: wait for producer's step t, and ring slot free
    if (tid == 0) {
      if (L > 0)                    spin_ge(&prog[L-1], t + 1);
      if (out_ring && t >= wsteps)  spin_ge(&prog[L+1], t - wsteps + 1);
    }
    __syncthreads();

    // ---- Stage x_t into z (thread tid <-> batch row tid; rows are wave-private)
    {
      int b = tid;
      _Float16* zr = zbuf + b * 32;
      if (in_ring == nullptr) {
        const float* xr = x0 + ((size_t)t * BATCH + b) * 2;
        zr[0] = (_Float16)xr[0];
        zr[1] = (_Float16)xr[1];
      } else {
        const _Float16* ir = in_ring + ((size_t)(t & wmask) * BATCH + b) * Hin;
        for (int k = 0; k < Hin; ++k) {
          float v = (float)ir[k];
          if (relu_in) v = fmaxf(v, 0.0f);
          zr[k] = (_Float16)v;
        }
      }
    }
    __syncthreads();

    // ---- gates = [x|h] @ [Wih;Whh]^T via WMMA f16 (K = 32)
    v8f acc[2][NT];
    #pragma unroll
    for (int m = 0; m < 2; ++m) {
      int mt = wave * 2 + m;
      const _Float16* zr = zbuf + (size_t)(mt * 16 + n) * 32;
      V16 af;
      af.h2[0] = *(const v8h*)(zr + klo);       // K = klo..klo+7
      af.h2[1] = *(const v8h*)(zr + 16 + klo);  // K = 16+klo..16+klo+7
      #pragma unroll
      for (int nt = 0; nt < NT; ++nt) {
        v8f cz = {};
        acc[m][nt] = __builtin_amdgcn_wmma_f32_16x16x32_f16(
            false, af.v, false, bfr[nt].v, (short)0, cz, false, false);
      }
    }

    // ---- Pointwise LSTM update fully in registers (gate gather via shfl_xor)
    int slot = t & wmask;
    #pragma unroll
    for (int m = 0; m < 2; ++m) {
      int mt = wave * 2 + m;
      #pragma unroll
      for (int r = 0; r < 8; ++r) {
        float gi, gf, gg, go;
        if constexpr (H == 16) {
          gi = acc[m][0][r]; gf = acc[m][1][r]; gg = acc[m][2][r]; go = acc[m][3][r];
        } else if constexpr (H == 8) {
          float a0 = acc[m][0][r], a1 = acc[m][1][r];
          gf = __shfl_xor(a0, 8, 32);
          go = __shfl_xor(a1, 8, 32);
          gi = a0; gg = a1;
        } else {
          float a0 = acc[m][0][r];
          gf = __shfl_xor(a0, 4, 32);
          gg = __shfl_xor(a0, 8, 32);
          go = __shfl_xor(a0, 12, 32);
          gi = a0;
        }
        float cn = sigm(gf + Bf) * creg[m][r] + sigm(gi + Bi) * tanh_f(gg + Bg);
        creg[m][r] = cn;
        float hv = sigm(go + Bo) * tanh_f(cn);
        if (n < H) {                               // owner lane for (b, hh=n)
          int b = mt * 16 + r + (hig ? 8 : 0);
          zbuf[b * 32 + Din + n] = (_Float16)hv;   // recurrent h for t+1
          if (out_ring)
            out_ring[((size_t)slot * BATCH + b) * H + n] = (_Float16)hv;
        }
      }
    }

    // ---- Last layer: fused ReLU + FC (16 -> 4) straight to d_out
    if (out_ring == nullptr) {
      __syncthreads();
      int b = tid;
      const _Float16* hr = zbuf + b * 32 + Din;  // Din == 16 here
      float y0 = fcls[64], y1 = fcls[65], y2 = fcls[66], y3 = fcls[67];
      #pragma unroll
      for (int k = 0; k < 16; ++k) {
        float hv = fmaxf((float)hr[k], 0.0f);
        y0 += hv * fcls[0 * 16 + k];
        y1 += hv * fcls[1 * 16 + k];
        y2 += hv * fcls[2 * 16 + k];
        y3 += hv * fcls[3 * 16 + k];
      }
      float* orow = out_final + ((size_t)t * BATCH + b) * 4;
      orow[0] = y0; orow[1] = y1; orow[2] = y2; orow[3] = y3;
    }

    // ---- Publish step t (release so consumer's acquire sees ring data)
    __threadfence();
    __syncthreads();
    if (tid == 0)
      __hip_atomic_store(&prog[L], t + 1, __ATOMIC_RELEASE, __HIP_MEMORY_SCOPE_AGENT);
  }
}

__global__ void lstm_init(int* prog) {
  int t = threadIdx.x;
  if (t < NLAYER) prog[t] = 0;
}

__global__ void __launch_bounds__(256)
lstm_pipeline(const float* x,
  const float* l1_Wih0, const float* l1_Whh0, const float* l1_bih0, const float* l1_bhh0,
  const float* l1_Wih,  const float* l1_Whh,  const float* l1_bih,  const float* l1_bhh,
  const float* l2_Wih0, const float* l2_Whh0, const float* l2_bih0, const float* l2_bhh0,
  const float* l2_Wih,  const float* l2_Whh,  const float* l2_bih,  const float* l2_bhh,
  const float* l3_Wih0, const float* l3_Whh0, const float* l3_bih0, const float* l3_bhh0,
  const float* l3_Wih,  const float* l3_Whh,  const float* l3_bih,  const float* l3_bhh,
  const float* fc_W, const float* fc_b,
  float* out, int* prog, _Float16* rings, int wsteps)
{
  __shared__ __align__(16) _Float16 zbuf[BATCH * 32];
  __shared__ float fcls[68];
  const int tid = threadIdx.x;
  if (tid < 68) fcls[tid] = (tid < 64) ? fc_W[tid] : fc_b[tid - 64];

  int L = blockIdx.x;
  const float *Wih, *Whh, *bihp, *bhhp;
  int Din; bool relu_in = false;

  if (L < 32) {
    if (L == 0) { Wih = l1_Wih0; Whh = l1_Whh0; bihp = l1_bih0; bhhp = l1_bhh0; Din = 2; }
    else { int i = L - 1;
      Wih = l1_Wih + (size_t)i*16*4; Whh = l1_Whh + (size_t)i*16*4;
      bihp = l1_bih + (size_t)i*16;  bhhp = l1_bhh + (size_t)i*16; Din = 4; }
  } else if (L < 96) {
    int k = L - 32;
    if (k == 0) { Wih = l2_Wih0; Whh = l2_Whh0; bihp = l2_bih0; bhhp = l2_bhh0; Din = 4; relu_in = true; }
    else { int i = k - 1;
      Wih = l2_Wih + (size_t)i*32*8; Whh = l2_Whh + (size_t)i*32*8;
      bihp = l2_bih + (size_t)i*32;  bhhp = l2_bhh + (size_t)i*32; Din = 8; }
  } else {
    int k = L - 96;
    if (k == 0) { Wih = l3_Wih0; Whh = l3_Whh0; bihp = l3_bih0; bhhp = l3_bhh0; Din = 8; relu_in = true; }
    else { int i = k - 1;
      Wih = l3_Wih + (size_t)i*64*16; Whh = l3_Whh + (size_t)i*64*16;
      bihp = l3_bih + (size_t)i*64;   bhhp = l3_bhh + (size_t)i*64; Din = 16; }
  }
  int Hin = (L == 0) ? 2 : ((L - 1) < 32 ? 4 : ((L - 1) < 96 ? 8 : 16));

  const size_t ring_stride = (size_t)wsteps * BATCH * 16;  // capacity for H<=16
  const _Float16* in_ring = (L == 0) ? nullptr : rings + (size_t)(L - 1) * ring_stride;
  _Float16* out_ring = (L == NLAYER - 1) ? nullptr : rings + (size_t)L * ring_stride;

  if (L < 32)
    run_layer<4 >(Wih, Whh, bihp, bhhp, Din, Hin, relu_in, x, in_ring, out_ring, out, fcls, zbuf, prog, L, wsteps);
  else if (L < 96)
    run_layer<8 >(Wih, Whh, bihp, bhhp, Din, Hin, relu_in, x, in_ring, out_ring, out, fcls, zbuf, prog, L, wsteps);
  else
    run_layer<16>(Wih, Whh, bihp, bhhp, Din, Hin, relu_in, x, in_ring, out_ring, out, fcls, zbuf, prog, L, wsteps);
}

extern "C" void kernel_launch(void* const* d_in, const int* in_sizes, int n_in,
                              void* d_out, int out_size, void* d_ws, size_t ws_size,
                              hipStream_t stream) {
  (void)in_sizes; (void)n_in; (void)out_size;
  const float* p[27];
  for (int i = 0; i < 27; ++i) p[i] = (const float*)d_in[i];

  int* prog = (int*)d_ws;
  _Float16* rings = (_Float16*)((char*)d_ws + 1024);

  // Ring depth: largest power of two (<=16) that fits in the workspace.
  int wsteps = 16;
  while (wsteps > 1 &&
         (size_t)1024 + (size_t)(NLAYER - 1) * wsteps * BATCH * 16 * 2 > ws_size)
    wsteps >>= 1;

  lstm_init<<<1, 256, 0, stream>>>(prog);
  lstm_pipeline<<<NLAYER, 256, 0, stream>>>(
      p[0],
      p[1],  p[2],  p[3],  p[4],  p[5],  p[6],  p[7],  p[8],
      p[9],  p[10], p[11], p[12], p[13], p[14], p[15], p[16],
      p[17], p[18], p[19], p[20], p[21], p[22], p[23], p[24],
      p[25], p[26],
      (float*)d_out, prog, rings, wsteps);
}
